// StackedBidirectionalLSTMEncoder_79783312491109
// MI455X (gfx1250) — compile-verified, matrix operand order checked by reading, not run on hardware
//
#include <hip/hip_runtime.h>
#include <stdint.h>

// ---------------------------------------------------------------------------
// Stacked bidirectional LSTM encoder for gfx1250 (MI455X).
// B=64, T=256, V=128, E=512, U=512.
// bf16 WMMA (v_wmma_f32_16x16x32_bf16) with f32 accumulate.
// Persistent cooperative scan: 32 WGs/direction, per-step device barrier.
// Both W and Ur slices staged in LDS (128/192 KB dynamic) in B-fragment form.
// Software-pipelined: next step's x@W GEMM overlaps the barrier wait.
// ---------------------------------------------------------------------------

typedef __attribute__((ext_vector_type(16))) __bf16 v16bf;
typedef __attribute__((ext_vector_type(8)))  float  v8f;

#define B_  64
#define T_  256
#define E_  512
#define U_  512
#define N4U 2048   // 4*U

__device__ __forceinline__ unsigned short f2bf(float f) {
  unsigned u = __float_as_uint(f);
  u += 0x7fffu + ((u >> 16) & 1u);           // round-to-nearest-even
  return (unsigned short)(u >> 16);
}

union FragBF { uint4 u4[2]; v16bf v; };

__device__ __forceinline__ v8f wmma_bf16(const FragBF& a, const FragBF& b, v8f c) {
  return __builtin_amdgcn_wmma_f32_16x16x32_bf16(false, a.v, false, b.v,
                                                 (short)0, c, false, false);
}

__device__ __forceinline__ float sigmoidf_(float x) {
  return 1.0f / (1.0f + __expf(-x));
}

// ---------------------------------------------------------------------------
// Kernel 1: embedding gather + mask.  xbf: time-major [T*B, E] bf16.
// ---------------------------------------------------------------------------
__global__ void embed_mask_kernel(const int* __restrict__ enc,
                                  const float* __restrict__ emb,
                                  unsigned short* __restrict__ xbf,
                                  unsigned char* __restrict__ mask) {
  int row = blockIdx.x;                       // t*64 + b
  int t = row >> 6, b = row & 63;
  int tok = enc[b * T_ + t];
  if (threadIdx.x == 0) mask[row] = (tok != 0) ? 1 : 0;
  const float* src = emb + (size_t)tok * E_;
  unsigned short* dst = xbf + (size_t)row * E_;
  for (int e = threadIdx.x; e < E_; e += blockDim.x) dst[e] = f2bf(src[e]);
}

// ---------------------------------------------------------------------------
// Kernel 2: weight swizzle f32[K][2048] -> bf16 WMMA B-fragment layout.
//   k = kt*32 + (i>>3)*16 + ((lane>>4)<<3) + (i&7),  n = nt*16 + (lane&15)
// ---------------------------------------------------------------------------
__global__ void swizzle_kernel(const float* __restrict__ W,
                               unsigned short* __restrict__ out, int K) {
  int KT = K >> 5;
  int o = blockIdx.x * blockDim.x + threadIdx.x;   // < K*2048
  int li   = o & 511;
  int lane = li >> 4;
  int i    = li & 15;
  int ktnt = o >> 9;
  int kt = ktnt % KT;
  int nt = ktnt / KT;
  int k = kt * 32 + ((i >> 3) << 4) + ((lane >> 4) << 3) + (i & 7);
  int n = nt * 16 + (lane & 15);
  out[o] = f2bf(W[(size_t)k * N4U + n]);
}

// ---------------------------------------------------------------------------
// Kernel 3: zero h ping-pong buffers and per-step barrier counters.
// ---------------------------------------------------------------------------
__global__ void init_state_kernel(unsigned short* __restrict__ hbuf,
                                  int* __restrict__ ctrs) {
  int tid = blockIdx.x * blockDim.x + threadIdx.x;
  int stride = gridDim.x * blockDim.x;
  const int total_h = 2 * 2 * B_ * U_;
  for (int i = tid; i < total_h; i += stride) hbuf[i] = 0;
  for (int i = tid; i < 2 * T_; i += stride) ctrs[i] = 0;
}

// ---------------------------------------------------------------------------
// Kernel 4: persistent bidirectional LSTM scan (software-pipelined).
// grid = 64 WGs (dir = blkIdx>>5, g = blkIdx&31 -> hidden cols u0=g*16..+16)
// block = 128 threads = 4 waves; wave w owns batch rows w*16..+16.
// Dynamic LDS: ldsU[4096 uint4] (Ur slice) + ldsW[KTx*256 uint4] (W slice).
// ---------------------------------------------------------------------------
__global__ __launch_bounds__(128, 1)
void lstm_scan_kernel(const unsigned short* __restrict__ xbf, int Kx,
                      const unsigned short* __restrict__ Wf_s,
                      const unsigned short* __restrict__ Wb_s,
                      const unsigned short* __restrict__ Uf_s,
                      const unsigned short* __restrict__ Ub_s,
                      const float* __restrict__ bias_f,
                      const float* __restrict__ bias_b,
                      const unsigned char* __restrict__ mask,
                      unsigned short* __restrict__ hbuf,
                      int* __restrict__ ctrs,
                      int mode,                      // 0: ys->bf16 x2, 1: f32 out
                      unsigned short* __restrict__ ys_bf,
                      float* __restrict__ out_f,
                      float* __restrict__ h2,
                      float* __restrict__ c2) {
  const int dir = blockIdx.x >> 5;                 // 0=fwd, 1=bwd
  const int g   = blockIdx.x & 31;                 // hidden-column chunk
  const int u0  = g * 16;
  const int mtile = threadIdx.x >> 5;              // wave id = batch tile
  const int lane  = threadIdx.x & 31;
  const int lhalf = lane >> 4;
  const int Mloc  = lane & 15;
  const int ncol  = lane & 15;                     // D-layout N index

  const unsigned short* Ws = dir ? Wb_s : Wf_s;
  const unsigned short* Us = dir ? Ub_s : Uf_s;
  const float*          bs = dir ? bias_b : bias_f;
  const uint4* W4 = (const uint4*)Ws;
  const uint4* U4 = (const uint4*)Us;
  const int KTx = Kx >> 5;                         // 16 or 32
  const int KTu = U_ >> 5;                         // 16

  extern __shared__ uint4 lds[];
  uint4* ldsU = lds;                               // [q][kt][lane][2] : 4096
  uint4* ldsW = lds + 4096;                        // [q][kt][lane][2] : KTx*256

  // --- stage Ur slice into LDS (fragment layout) ---
  for (int idx = threadIdx.x; idx < 4096; idx += 128) {
    int q  = idx >> 10;
    int kt = (idx >> 6) & 15;
    int r2 = idx & 63;                             // lane*2 + half
    ldsU[idx] = U4[((size_t)(q * 32 + g) * KTu + kt) * 64 + r2];
  }
  // --- stage W slice into LDS (fragment layout) ---
  {
    const int wtot = KTx * 256;                    // uint4 count
    const int tilesz = KTx * 64;
    for (int idx = threadIdx.x; idx < wtot; idx += 128) {
      int q   = idx / tilesz;
      int rem = idx - q * tilesz;
      int kt  = rem >> 6;
      int r2  = rem & 63;
      ldsW[idx] = W4[((size_t)(q * 32 + g) * KTx + kt) * 64 + r2];
    }
  }
  __syncthreads();

  // persistent per-wave state (f32, D layout)
  v8f c_reg = {0.f, 0.f, 0.f, 0.f, 0.f, 0.f, 0.f, 0.f};
  v8f h_reg = {0.f, 0.f, 0.f, 0.f, 0.f, 0.f, 0.f, 0.f};

  const float bi_ = bs[0 * U_ + u0 + ncol];
  const float bf_ = bs[1 * U_ + u0 + ncol];
  const float bg_ = bs[2 * U_ + u0 + ncol];
  const float bo_ = bs[3 * U_ + u0 + ncol];

  const int arow = mtile * 16 + Mloc;              // A-operand row (batch)

  // x-projection GEMM for timestep t into xacc (A from global, B from LDS)
  auto compute_xpart = [&](int t, v8f* xacc) {
    const unsigned short* xrow = xbf + (size_t)(t * B_ + arow) * Kx;
#pragma unroll 4
    for (int kt = 0; kt < KTx; ++kt) {
      FragBF a;
      const unsigned short* ap = xrow + kt * 32 + lhalf * 8;
      a.u4[0] = *(const uint4*)(ap);
      a.u4[1] = *(const uint4*)(ap + 16);
#pragma unroll
      for (int q = 0; q < 4; ++q) {
        FragBF b;
        int li = ((q * KTx + kt) * 32 + lane) * 2;
        b.u4[0] = ldsW[li];
        b.u4[1] = ldsW[li + 1];
        xacc[q] = wmma_bf16(a, b, xacc[q]);
      }
    }
  };

  // prologue: x-projection for the first step
  v8f xacc[4];
#pragma unroll
  for (int q = 0; q < 4; ++q)
    xacc[q] = (v8f){0.f, 0.f, 0.f, 0.f, 0.f, 0.f, 0.f, 0.f};
  compute_xpart(dir ? (T_ - 1) : 0, xacc);

#pragma unroll 1
  for (int s = 0; s < T_; ++s) {
    const int t = dir ? (T_ - 1 - s) : s;

    // ---- wait for previous step's h to be published (overlapped by the
    //      x-projection done at the end of the previous iteration) ----
    if (s > 0) {
      if (threadIdx.x == 0) {
        int* c = &ctrs[dir * T_ + (s - 1)];
        while (__hip_atomic_load(c, __ATOMIC_ACQUIRE, __HIP_MEMORY_SCOPE_AGENT) < 32)
          __builtin_amdgcn_s_sleep(1);
      }
      __syncthreads();
      __threadfence();                             // acquire peers' h writes
    }

    v8f acc[4];
#pragma unroll
    for (int q = 0; q < 4; ++q) acc[q] = xacc[q];

    // ---- z += h_{t-1} @ Ur (K = 512), A from global h buffer, B from LDS ----
    {
      const unsigned short* hread =
          hbuf + ((size_t)dir * 2 + (s & 1)) * (B_ * U_) + (size_t)arow * U_;
#pragma unroll 4
      for (int kt = 0; kt < KTu; ++kt) {
        FragBF a;
        const unsigned short* ap = hread + kt * 32 + lhalf * 8;
        a.u4[0] = *(const uint4*)(ap);
        a.u4[1] = *(const uint4*)(ap + 16);
#pragma unroll
        for (int q = 0; q < 4; ++q) {
          FragBF b;
          int li = ((q * 16 + kt) * 32 + lane) * 2;
          b.u4[0] = ldsU[li];
          b.u4[1] = ldsU[li + 1];
          acc[q] = wmma_bf16(a, b, acc[q]);
        }
      }
    }

    // ---- gates + state update (Keras mask semantics: carry when masked) ----
    const int mbase = mtile * 16 + lhalf * 8;      // first of 8 D-layout rows
    uint64_t m8 = *(const uint64_t*)(mask + t * B_ + mbase);

    unsigned short* hnext =
        hbuf + ((size_t)dir * 2 + ((s + 1) & 1)) * (B_ * U_);

#pragma unroll
    for (int r = 0; r < 8; ++r) {
      float iv = sigmoidf_(acc[0][r] + bi_);
      float fv = sigmoidf_(acc[1][r] + bf_);
      float gv = tanhf(acc[2][r] + bg_);
      float ov = sigmoidf_(acc[3][r] + bo_);
      float cn = fv * c_reg[r] + iv * gv;
      float hn = ov * tanhf(cn);
      bool m = ((m8 >> (8 * r)) & 1ull) != 0ull;
      c_reg[r] = m ? cn : c_reg[r];
      h_reg[r] = m ? hn : h_reg[r];

      int Mg  = mbase + r;
      int col = u0 + ncol;
      unsigned short hb = f2bf(h_reg[r]);
      hnext[(size_t)Mg * U_ + col] = hb;
      if (mode == 0) {
        ys_bf[(size_t)(t * B_ + Mg) * (2 * U_) + dir * U_ + col] = hb;
      } else {
        out_f[((size_t)Mg * T_ + t) * (2 * U_) + dir * U_ + col] = h_reg[r];
      }
    }

    // ---- signal barrier for step s as early as possible ----
    __threadfence();                               // release h writes
    __syncthreads();
    if (threadIdx.x == 0) {
      __hip_atomic_fetch_add(&ctrs[dir * T_ + s], 1, __ATOMIC_ACQ_REL,
                             __HIP_MEMORY_SCOPE_AGENT);
    }

    // ---- overlap: x-projection GEMM for the NEXT step hides barrier wait ----
    if (s + 1 < T_) {
#pragma unroll
      for (int q = 0; q < 4; ++q)
        xacc[q] = (v8f){0.f, 0.f, 0.f, 0.f, 0.f, 0.f, 0.f, 0.f};
      compute_xpart(dir ? (T_ - 2 - s) : (s + 1), xacc);
    }
  }

  // ---- final carry states (layer 2 only): h2/c2 [B, 2U] f32 ----
  if (mode == 1) {
#pragma unroll
    for (int r = 0; r < 8; ++r) {
      int Mg  = mtile * 16 + lhalf * 8 + r;
      int col = u0 + ncol;
      h2[(size_t)Mg * (2 * U_) + dir * U_ + col] = h_reg[r];
      c2[(size_t)Mg * (2 * U_) + dir * U_ + col] = c_reg[r];
    }
  }
}

// ---------------------------------------------------------------------------
// Host launcher
// ---------------------------------------------------------------------------
extern "C" void kernel_launch(void* const* d_in, const int* in_sizes, int n_in,
                              void* d_out, int out_size, void* d_ws, size_t ws_size,
                              hipStream_t stream) {
  (void)in_sizes; (void)n_in; (void)out_size; (void)ws_size;

  const int*   enc = (const int*)d_in[0];
  const float* emb = (const float*)d_in[1];
  const float* W1f = (const float*)d_in[2];
  const float* U1f = (const float*)d_in[3];
  const float* b1f = (const float*)d_in[4];
  const float* W1b = (const float*)d_in[5];
  const float* U1b = (const float*)d_in[6];
  const float* b1b = (const float*)d_in[7];
  const float* W2f = (const float*)d_in[8];
  const float* U2f = (const float*)d_in[9];
  const float* b2f = (const float*)d_in[10];
  const float* W2b = (const float*)d_in[11];
  const float* U2b = (const float*)d_in[12];
  const float* b2b = (const float*)d_in[13];
  float* out = (float*)d_out;

  // workspace layout (bytes)
  char* ws = (char*)d_ws;
  size_t off = 0;
  unsigned short* xbf  = (unsigned short*)(ws + off); off += (size_t)T_ * B_ * E_ * 2;
  unsigned short* x2bf = (unsigned short*)(ws + off); off += (size_t)T_ * B_ * 2 * U_ * 2;
  unsigned char*  mask = (unsigned char*) (ws + off); off += 65536;
  unsigned short* W1f_s = (unsigned short*)(ws + off); off += (size_t)E_ * N4U * 2;
  unsigned short* W1b_s = (unsigned short*)(ws + off); off += (size_t)E_ * N4U * 2;
  unsigned short* U1f_s = (unsigned short*)(ws + off); off += (size_t)U_ * N4U * 2;
  unsigned short* U1b_s = (unsigned short*)(ws + off); off += (size_t)U_ * N4U * 2;
  unsigned short* W2f_s = (unsigned short*)(ws + off); off += (size_t)(2 * U_) * N4U * 2;
  unsigned short* W2b_s = (unsigned short*)(ws + off); off += (size_t)(2 * U_) * N4U * 2;
  unsigned short* U2f_s = (unsigned short*)(ws + off); off += (size_t)U_ * N4U * 2;
  unsigned short* U2b_s = (unsigned short*)(ws + off); off += (size_t)U_ * N4U * 2;
  unsigned short* hbuf  = (unsigned short*)(ws + off); off += (size_t)2 * 2 * B_ * U_ * 2;
  int*            ctrs  = (int*)(ws + off);            off += (size_t)2 * T_ * 4;

  // raise dynamic-LDS cap for the scan kernel (host-side, capture-safe)
  const int shmem1 = (4096 + 16 * 256) * 16;   // 128 KB (layer 1, KTx=16)
  const int shmem2 = (4096 + 32 * 256) * 16;   // 192 KB (layer 2, KTx=32)
  hipFuncSetAttribute((const void*)lstm_scan_kernel,
                      hipFuncAttributeMaxDynamicSharedMemorySize, shmem2);

  // 1) embedding + mask
  embed_mask_kernel<<<T_ * B_, 128, 0, stream>>>(enc, emb, xbf, mask);

  // 2) swizzle all weights to WMMA B-fragment bf16 layout
  swizzle_kernel<<<(E_ * N4U) / 256, 256, 0, stream>>>(W1f, W1f_s, E_);
  swizzle_kernel<<<(E_ * N4U) / 256, 256, 0, stream>>>(W1b, W1b_s, E_);
  swizzle_kernel<<<(U_ * N4U) / 256, 256, 0, stream>>>(U1f, U1f_s, U_);
  swizzle_kernel<<<(U_ * N4U) / 256, 256, 0, stream>>>(U1b, U1b_s, U_);
  swizzle_kernel<<<(2 * U_ * N4U) / 256, 256, 0, stream>>>(W2f, W2f_s, 2 * U_);
  swizzle_kernel<<<(2 * U_ * N4U) / 256, 256, 0, stream>>>(W2b, W2b_s, 2 * U_);
  swizzle_kernel<<<(U_ * N4U) / 256, 256, 0, stream>>>(U2f, U2f_s, U_);
  swizzle_kernel<<<(U_ * N4U) / 256, 256, 0, stream>>>(U2b, U2b_s, U_);

  // 3) layer 1 scan (both directions concurrently)
  init_state_kernel<<<64, 256, 0, stream>>>(hbuf, ctrs);
  lstm_scan_kernel<<<64, 128, shmem1, stream>>>(xbf, E_,
                                                W1f_s, W1b_s, U1f_s, U1b_s,
                                                b1f, b1b, mask, hbuf, ctrs,
                                                /*mode=*/0, x2bf,
                                                nullptr, nullptr, nullptr);

  // 4) layer 2 scan -> out2, h2, c2
  const size_t O1 = (size_t)B_ * T_ * 2 * U_;
  const size_t O2 = O1 + (size_t)B_ * 2 * U_;
  init_state_kernel<<<64, 256, 0, stream>>>(hbuf, ctrs);
  lstm_scan_kernel<<<64, 128, shmem2, stream>>>(x2bf, 2 * U_,
                                                W2f_s, W2b_s, U2f_s, U2b_s,
                                                b2f, b2b, mask, hbuf, ctrs,
                                                /*mode=*/1, nullptr,
                                                out, out + O1, out + O2);
}
